// PointNetpp_msg_21586505630015
// MI455X (gfx1250) — matrix operand order repaired and checked by reading, not compiled, not run on hardware
//
#include <hip/hip_runtime.h>
#include <hip/hip_bf16.h>
#include <cstdint>

typedef __attribute__((ext_vector_type(16))) _Float16 v16h;
typedef __attribute__((ext_vector_type(8)))  float    v8f;
typedef __attribute__((ext_vector_type(4)))  unsigned int u32x4;
typedef __attribute__((ext_vector_type(8)))  unsigned int u32x8;

#define NTOT 16384
#define NC 4096
#define NPOINT 2050
#define SALL 8200
#define OUT_XY 0
#define OUT_XYZ 16400
#define OUT_PTS 41000
#define OUT_SPLIT 5289000

// B-operand (and activation-buffer) layout: 32(K)x16(N) f16 blocks, 512 f16 each.
// Within a block: lane = (k%32 >= 16 ? 16:0) + (col%16), t = k%16 -> f16 index lane*16+t.
// Contiguous k order (k = 16*half + t), same as the A layout: one contiguous 32-byte load per lane.
__device__ __forceinline__ size_t blayout(int k, int col, int nKB) {
  int kb = k >> 5, kk = k & 31;
  int khalf = kk >> 4, t = kk & 15;
  int ct = col >> 4, n = col & 15;
  int lane = (khalf << 4) + n;
  return ((size_t)(ct * nKB + kb) << 9) + (size_t)(lane << 4) + (size_t)t;
}

// ---------------- FPS: one workgroup per chunk, LDS-resident point cloud ----------
__global__ __launch_bounds__(1024) void fps_kernel(const float* __restrict__ xyz,
                                                   int base, int* __restrict__ fidx) {
  extern __shared__ float smem[];
  float* px = smem;
  float* py = smem + NC;
  float* pz = smem + 2 * NC;
  float* dist = smem + 3 * NC;
  float* rv = smem + 4 * NC;
  int*   ri = (int*)(smem + 4 * NC + 1024);
  __shared__ int s_far;
  int tid = threadIdx.x;
  for (int i = tid; i < NC; i += 1024) {
    px[i] = xyz[base + i];
    py[i] = xyz[NTOT + base + i];
    pz[i] = xyz[2 * NTOT + base + i];
    dist[i] = 1e10f;
  }
  if (tid == 0) s_far = 0;
  __syncthreads();
  for (int it = 0; it < NPOINT; ++it) {
    int far = s_far;
    if (tid == 0) fidx[it] = far;
    float cx = px[far], cy = py[far], cz = pz[far];
    float bv = -1.0f; int bi = 0;
    for (int i = tid; i < NC; i += 1024) {
      float dx = px[i] - cx, dy = py[i] - cy, dz = pz[i] - cz;
      float d = dx * dx + dy * dy + dz * dz;
      float nd = fminf(dist[i], d);
      dist[i] = nd;
      if (nd > bv) { bv = nd; bi = i; }   // strict > keeps first index (argmax tie rule)
    }
    rv[tid] = bv; ri[tid] = bi;
    __syncthreads();
    for (int st = 512; st > 0; st >>= 1) {
      if (tid < st) {
        float v2 = rv[tid + st]; int i2 = ri[tid + st];
        if (v2 > rv[tid] || (v2 == rv[tid] && i2 < ri[tid])) { rv[tid] = v2; ri[tid] = i2; }
      }
      __syncthreads();
    }
    if (tid == 0) s_far = ri[0];
    __syncthreads();
  }
}

// ---------------- gather centroid coords into output -----------------------------
__global__ void gather_kernel(const float* __restrict__ xy, const float* __restrict__ xyz,
                              const int* __restrict__ fidx, float* __restrict__ out,
                              int base, int sg) {
  int s = blockIdx.x * blockDim.x + threadIdx.x;
  if (s >= NPOINT) return;
  int j = fidx[s];
  out[OUT_XY  + 0 * SALL + sg + s] = xy[0 * NTOT + base + j];
  out[OUT_XY  + 1 * SALL + sg + s] = xy[1 * NTOT + base + j];
  out[OUT_XYZ + 0 * SALL + sg + s] = xyz[0 * NTOT + base + j];
  out[OUT_XYZ + 1 * SALL + sg + s] = xyz[1 * NTOT + base + j];
  out[OUT_XYZ + 2 * SALL + sg + s] = xyz[2 * NTOT + base + j];
}

// ---------------- ball query: first K indices (ascending) within radius ----------
__global__ __launch_bounds__(256) void query_ball_kernel(const float* __restrict__ xyz, int base,
                                                         const float* __restrict__ newxyz, int sg,
                                                         float r2, int K, int* __restrict__ gi) {
  __shared__ float tx[512], ty[512], tz[512];
  int s = blockIdx.x * blockDim.x + threadIdx.x;
  bool active = s < NPOINT;
  float nx = 0.f, ny = 0.f, nz = 0.f;
  if (active) {
    nx = newxyz[0 * SALL + sg + s];
    ny = newxyz[1 * SALL + sg + s];
    nz = newxyz[2 * SALL + sg + s];
  }
  int cnt = 0;
  int* outp = gi + (size_t)s * K;
  for (int tile = 0; tile < NC / 512; ++tile) {
    __syncthreads();
    for (int i = threadIdx.x; i < 512; i += 256) {
      tx[i] = xyz[0 * NTOT + base + tile * 512 + i];
      ty[i] = xyz[1 * NTOT + base + tile * 512 + i];
      tz[i] = xyz[2 * NTOT + base + tile * 512 + i];
    }
    __syncthreads();
    if (active && cnt < K) {
      for (int i = 0; i < 512 && cnt < K; ++i) {
        float dx = tx[i] - nx, dy = ty[i] - ny, dz = tz[i] - nz;
        if (dx * dx + dy * dy + dz * dz <= r2) outp[cnt++] = tile * 512 + i;
      }
    }
  }
  if (active) {
    int first = (cnt > 0) ? outp[0] : 0;  // centroid always matches itself -> cnt>=1
    for (int i = cnt; i < K; ++i) outp[i] = first;
  }
}

// ---------------- grouping: build layer-1 input, f16, B-operand layout ------------
__global__ void group_kernel(const float* __restrict__ xyz, const float* __restrict__ points,
                             const int* __restrict__ gi, const float* __restrict__ newxyz,
                             int base, int K, int sg, _Float16* __restrict__ X, int nKB) {
  int col = blockIdx.x * blockDim.x + threadIdx.x;
  int cols = K * NPOINT;
  if (col >= cols) return;
  int s = col / K;
  int j = gi[col];
  for (int ch = 0; ch < 64; ++ch)
    X[blayout(ch, col, nKB)] = (_Float16)points[ch * NTOT + base + j];
  for (int d = 0; d < 3; ++d) {
    float v = xyz[d * NTOT + base + j] - newxyz[d * SALL + sg + s];
    X[blayout(64 + d, col, nKB)] = (_Float16)v;
  }
  for (int k = 67; k < nKB * 32; ++k)
    X[blayout(k, col, nKB)] = (_Float16)0.0f;   // zero-pad K dim to 96
}

// ---------------- weight staging into A-operand (16x32) layout --------------------
__global__ void prep_w_kernel(const float* __restrict__ W, _Float16* __restrict__ A,
                              int Cout, int Cin, int Kpad) {
  int idx = blockIdx.x * blockDim.x + threadIdx.x;
  if (idx >= Cout * Kpad) return;
  int m = idx / Kpad, k = idx % Kpad;
  float v = (k < Cin) ? W[m * Cin + k] : 0.0f;
  int nKB = Kpad >> 5;
  int mt = m >> 4, mm = m & 15;
  int kb = k >> 5, kk = k & 31;
  // A map (matches B layout): lanes 0-15 hold K=0..15, lanes 16-31 hold K=16..31
  int khalf = kk >> 4;
  int t = kk & 15;
  int lane = (khalf << 4) + mm;
  A[((size_t)(mt * nKB + kb) << 9) + (size_t)(lane << 4) + (size_t)t] = (_Float16)v;
}

// ---------------- GEMM: 4 waves/block, B tile staged to LDS via TDM ---------------
// Block = (ct, 4 M-tiles). Wave 0 issues one tensor_load_to_lds for the whole B
// column tile (nKB * 1KB, contiguous), waits on TENSORcnt, then all 4 waves read B
// from LDS (ds_load_b128) while streaming their private A operand from global.
__global__ __launch_bounds__(128) void gemm_wmma_kernel(const _Float16* __restrict__ A,
    const _Float16* __restrict__ B, _Float16* __restrict__ Y,
    int nKB, int nKBo, int nMT, float* __restrict__ ssum, float* __restrict__ ssq) {
  __shared__ _Float16 tileB[4 * 512];               // up to nKB=4 blocks of 32x16 f16
  int ct = blockIdx.x;
  int wave = threadIdx.x >> 5;
  int lane = threadIdx.x & 31;
  if (wave == 0) {                                  // wave-uniform branch: TDM ignores EXEC
    unsigned long long gaddr =
        (unsigned long long)(uintptr_t)(B + (((size_t)ct * nKB) << 9));
    unsigned ldsoff =
        (unsigned)(unsigned long long)(__attribute__((address_space(3))) char*)tileB;
    unsigned e8 = (unsigned)((nKB << 10) >> 3);     // tile length in 8-byte elements
    // D# group 0: count=1 | lds_addr | global_addr[56:0] | type=2
    u32x4 g0 = { 1u, ldsoff, (unsigned)gaddr,
                 (unsigned)((gaddr >> 32) & 0x1FFFFFFull) | (2u << 30) };
    // D# group 1: data_size=8B, 1-D tile: tensor_dim0=tile_dim0=stride=e8
    u32x8 g1 = { 3u << 16,
                 (e8 & 0xFFFFu) << 16,
                 (e8 >> 16) & 0xFFFFu,
                 (e8 & 0xFFFFu) << 16,
                 0u, e8, 0u, 0u };
    asm volatile("tensor_load_to_lds %0, %1" :: "s"(g0), "s"(g1) : "memory");
    __builtin_amdgcn_s_wait_tensorcnt(0);
  }
  __syncthreads();
  int mt = blockIdx.y * 4 + wave;
  if (mt < nMT) {
    v8f c = {};
    const _Float16* Ap = A + (((size_t)mt * nKB) << 9) + (size_t)(lane << 4);
    const _Float16* Bl = tileB + (lane << 4);
    for (int kb = 0; kb < nKB; ++kb) {
      v16h a = *(const v16h*)(Ap + ((size_t)kb << 9));
      v16h b = *(const v16h*)(Bl + (kb << 9));
      c = __builtin_amdgcn_wmma_f32_16x16x32_f16(false, a, false, b, (short)0, c, false, false);
    }
    int hl = lane >> 4, n = lane & 15;
    int col = (ct << 4) + n;
    for (int r = 0; r < 8; ++r) {
      float v = c[r];
      int row = (mt << 4) + r + (hl << 3);          // D map: vgpr r = rows r / r+8
      Y[blayout(row, col, nKBo)] = (_Float16)v;     // pre-BN, B-layout for next layer
      float sv = v, sq = v * v;
      for (int off = 1; off < 16; off <<= 1) {      // reduce 16 cols within each half-wave
        sv += __shfl_xor(sv, off, 16);
        sq += __shfl_xor(sq, off, 16);
      }
      if (n == 0) { atomicAdd(&ssum[row], sv); atomicAdd(&ssq[row], sq); }
    }
  }
}

// ---------------- BN finalize + fused BN/ReLU in-place ----------------------------
__global__ void bn_finalize_kernel(const float* __restrict__ ssum, const float* __restrict__ ssq,
                                   const float* __restrict__ g, const float* __restrict__ be,
                                   float* __restrict__ a, float* __restrict__ b,
                                   int Cout, float invn) {
  int ch = threadIdx.x;
  if (ch >= Cout) return;
  float mu = ssum[ch] * invn;
  float var = ssq[ch] * invn - mu * mu;
  float sc = g[ch] * rsqrtf(var + 1e-5f);
  a[ch] = sc;
  b[ch] = be[ch] - mu * sc;   // conv bias absorbed by mean subtraction -> skipped
}

__global__ void bn_relu_kernel(_Float16* __restrict__ X, int nct, int nKB,
                               const float* __restrict__ a, const float* __restrict__ b) {
  size_t idx = (size_t)blockIdx.x * blockDim.x + threadIdx.x;
  size_t total = (size_t)nct * nKB * 512;
  if (idx >= total) return;
  int off = (int)(idx & 511);
  size_t blk = idx >> 9;
  int kb = (int)(blk % nKB);
  int lane = off >> 4, t = off & 15;
  int k = kb * 32 + ((lane >> 4) << 4) + t;  // invert blayout
  float v = (float)X[idx] * a[k] + b[k];
  X[idx] = (_Float16)fmaxf(v, 0.0f);
}

// ---------------- max over K samples -> output ------------------------------------
__global__ void maxpool_kernel(const _Float16* __restrict__ X, int nKB, int K,
                               int chOff, int sg, int Cout, float* __restrict__ out) {
  int idx = blockIdx.x * blockDim.x + threadIdx.x;
  if (idx >= Cout * NPOINT) return;
  int ch = idx / NPOINT, s = idx % NPOINT;
  float m = -1e30f;
  for (int kk = 0; kk < K; ++kk) {
    float v = (float)X[blayout(ch, s * K + kk, nKB)];
    m = fmaxf(m, v);
  }
  out[OUT_PTS + (size_t)(chOff + ch) * SALL + sg + s] = m;
}

__global__ void split_kernel(float* __restrict__ out) {
  int i = threadIdx.x;
  if (i < 5) out[OUT_SPLIT + i] = (float)(i * NPOINT);
}

// ==================================================================================
extern "C" void kernel_launch(void* const* d_in, const int* in_sizes, int n_in,
                              void* d_out, int out_size, void* d_ws, size_t ws_size,
                              hipStream_t stream) {
  (void)in_sizes; (void)n_in; (void)out_size; (void)ws_size;
  const float* xy = (const float*)d_in[0];
  const float* xyz = (const float*)d_in[1];
  const float* points = (const float*)d_in[2];
  float* out = (float*)d_out;

  // workspace carve-up (256B aligned regions)
  uintptr_t p = (uintptr_t)d_ws;
  auto take = [&](size_t bytes) { uintptr_t q = p; p = (p + bytes + 255) & ~(uintptr_t)255; return q; };
  float* ssum = (float*)take(256 * 4);
  float* ssq  = (float*)take(256 * 4);
  float* bn_a = (float*)take(256 * 4);
  float* bn_b = (float*)take(256 * 4);
  int* fidx   = (int*)take((size_t)NPOINT * 4);
  int* gi     = (int*)take((size_t)NPOINT * 64 * 4);
  _Float16* Wbuf = (_Float16*)take((size_t)3 * 32768 * 2);     // 3 layer slots, 64KB each
  size_t xelems = (size_t)8200 * 8 * 512;                      // max: 256ch x 131200 cols
  _Float16* XA = (_Float16*)take(xelems * 2);
  _Float16* XB = (_Float16*)take(xelems * 2);

  const int   Ks[3]     = {16, 32, 64};
  const float Rs[3]     = {0.2f, 0.4f, 0.8f};
  const int   MLP[3][3] = {{64, 64, 128}, {128, 128, 256}, {128, 128, 256}};
  const int   chOff[3]  = {0, 128, 384};

  for (int chunk = 0; chunk < 4; ++chunk) {
    int base = chunk * NC;
    int sg = chunk * NPOINT;
    size_t lds = (size_t)(4 * NC + 2 * 1024) * 4;
    hipLaunchKernelGGL(fps_kernel, dim3(1), dim3(1024), lds, stream, xyz, base, fidx);
    hipLaunchKernelGGL(gather_kernel, dim3((NPOINT + 255) / 256), dim3(256), 0, stream,
                       xy, xyz, fidx, out, base, sg);
    for (int sc = 0; sc < 3; ++sc) {
      int K = Ks[sc];
      int cols = K * NPOINT;
      int nct = cols / 16;
      float r2 = Rs[sc] * Rs[sc];
      int cin = 67;
      for (int l = 0; l < 3; ++l) {
        int cout = MLP[sc][l];
        int kpad = (l == 0) ? 96 : cin;
        const float* W = (const float*)d_in[4 + sc * 12 + l * 4 + 0];
        hipLaunchKernelGGL(prep_w_kernel, dim3((cout * kpad + 255) / 256), dim3(256), 0, stream,
                           W, Wbuf + l * 32768, cout, cin, kpad);
        cin = cout;
      }
      hipLaunchKernelGGL(query_ball_kernel, dim3((NPOINT + 255) / 256), dim3(256), 0, stream,
                         xyz, base, out + OUT_XYZ, sg, r2, K, gi);
      hipLaunchKernelGGL(group_kernel, dim3((cols + 127) / 128), dim3(128), 0, stream,
                         xyz, points, gi, out + OUT_XYZ, base, K, sg, XA, 3);
      _Float16* cur = XA; _Float16* nxt = XB;
      int nKBin = 3;
      for (int l = 0; l < 3; ++l) {
        int cout = MLP[sc][l];
        int nKBo = cout / 32;
        int nMT = cout / 16;
        hipMemsetAsync(ssum, 0, 256 * 4, stream);
        hipMemsetAsync(ssq, 0, 256 * 4, stream);
        hipLaunchKernelGGL(gemm_wmma_kernel, dim3(nct, cout / 64), dim3(128), 0, stream,
                           Wbuf + l * 32768, cur, nxt, nKBin, nKBo, nMT, ssum, ssq);
        const float* g  = (const float*)d_in[4 + sc * 12 + l * 4 + 2];
        const float* be = (const float*)d_in[4 + sc * 12 + l * 4 + 3];
        hipLaunchKernelGGL(bn_finalize_kernel, dim3(1), dim3(256), 0, stream,
                           ssum, ssq, g, be, bn_a, bn_b, cout, 1.0f / (float)cols);
        size_t tot = (size_t)nct * nKBo * 512;
        hipLaunchKernelGGL(bn_relu_kernel, dim3((unsigned)((tot + 255) / 256)), dim3(256), 0, stream,
                           nxt, nct, nKBo, bn_a, bn_b);
        _Float16* t = cur; cur = nxt; nxt = t;
        nKBin = nKBo;
      }
      int c3 = MLP[sc][2];
      hipLaunchKernelGGL(maxpool_kernel, dim3((c3 * NPOINT + 255) / 256), dim3(256), 0, stream,
                         cur, c3 / 32, K, chOff[sc], sg, c3, out);
    }
  }
  hipLaunchKernelGGL(split_kernel, dim3(1), dim3(8), 0, stream, out);
}
